// Qwen3MoeDecoderLayer_67851893342707
// MI455X (gfx1250) — matrix operand order repaired and checked by reading, not compiled, not run on hardware
//
#include <hip/hip_runtime.h>
#include <hip/hip_bf16.h>

// ---------------------------------------------------------------- constants
constexpr int TT  = 1024;   // tokens
constexpr int HH  = 1024;   // hidden
constexpr int NHD = 16;     // query heads
constexpr int NKV = 4;      // kv heads
constexpr int DD  = 64;     // head dim
constexpr int EE  = 32;     // experts
constexpr int TOPK= 4;
constexpr int II  = 512;    // expert intermediate
constexpr int QKVW = (NHD + 2*NKV) * DD;   // 1536
constexpr float EPSF = 1e-6f;
constexpr float THETAF = 1000000.0f;

typedef unsigned short u16;
typedef __attribute__((ext_vector_type(16))) __bf16 v16bf;
typedef __attribute__((ext_vector_type(8)))  float  v8f;

// ---------------------------------------------------------------- helpers
__device__ __forceinline__ u16 f2bf(float f) {
    unsigned u = __float_as_uint(f);
    u += 0x7fffu + ((u >> 16) & 1u);           // round-to-nearest-even
    return (u16)(u >> 16);
}
__device__ __forceinline__ float bf2f(u16 h) {
    return __uint_as_float(((unsigned)h) << 16);
}

union FragU { v16bf v; uint4 q[2]; };

// A-operand fragment (16x32 bf16): lane m=l&15, hi=l>>4.
// element e -> K = (e<8) ? 8*hi+e : 8+8*hi+e  => two 16B chunks at k+8*hi, k+16+8*hi
__device__ __forceinline__ v16bf load_fragA(const u16* __restrict__ row, int k, int hi) {
    FragU f;
    f.q[0] = *(const uint4*)(row + k + 8*hi);
    f.q[1] = *(const uint4*)(row + k + 16 + 8*hi);
    return f.v;
}
// B-operand fragment (32x16 bf16, stored as 16 rows of K): lane n=l&15, hi=l>>4.
// element e -> K = (e<8) ? 8*hi+e : 8+8*hi+e  => two 16B chunks at k+8*hi, k+16+8*hi
__device__ __forceinline__ v16bf load_fragB(const u16* __restrict__ row, int k, int hi) {
    FragU f;
    f.q[0] = *(const uint4*)(row + k + 8*hi);
    f.q[1] = *(const uint4*)(row + k + 16 + 8*hi);
    return f.v;
}

#define WMMA_BF16(a, b, c) \
    __builtin_amdgcn_wmma_f32_16x16x32_bf16(false, (a), false, (b), (short)0, (c), false, false)

// ---------------------------------------------------------------- f32 -> bf16 convert
__global__ void k_f2bf(const float* __restrict__ s, u16* __restrict__ d, int n) {
    int i = blockIdx.x * blockDim.x + threadIdx.x;
    int stride = gridDim.x * blockDim.x;
    for (; i < n; i += stride) d[i] = f2bf(s[i]);
}

// ---------------------------------------------------------------- prep1: x1 = hs+res, h1 = rms(x1)*ln1
__global__ void k_prep1(const float* __restrict__ hs, const float* __restrict__ res,
                        const float* __restrict__ ln1w,
                        float* __restrict__ x1f, u16* __restrict__ h1b) {
    int t = blockIdx.x, tid = threadIdx.x;
    __shared__ float sred[8];
    float v[4]; float ss = 0.f;
#pragma unroll
    for (int ii = 0; ii < 4; ++ii) {
        int i = tid + 256*ii;
        float x = hs[(size_t)t*HH + i] + res[(size_t)t*HH + i];
        v[ii] = x; x1f[(size_t)t*HH + i] = x; ss += x*x;
    }
#pragma unroll
    for (int off = 16; off >= 1; off >>= 1) ss += __shfl_xor(ss, off, 32);
    if ((tid & 31) == 0) sred[tid >> 5] = ss;
    __syncthreads();
    float tot = 0.f;
#pragma unroll
    for (int w = 0; w < 8; ++w) tot += sred[w];
    float sc = rsqrtf(tot / (float)HH + EPSF);
#pragma unroll
    for (int ii = 0; ii < 4; ++ii) {
        int i = tid + 256*ii;
        h1b[(size_t)t*HH + i] = f2bf(v[ii] * sc * ln1w[i]);
    }
}

// ---------------------------------------------------------------- prep2: res_out = attn+x1, zero moe, h2, zero cnt
__global__ void k_prep2(const float* __restrict__ attnf, const float* __restrict__ x1f,
                        const float* __restrict__ ln2w,
                        float* __restrict__ out_moe, float* __restrict__ out_res,
                        u16* __restrict__ h2b, int* __restrict__ cnt) {
    int t = blockIdx.x, tid = threadIdx.x;
    __shared__ float sred[8];
    float v[4]; float ss = 0.f;
#pragma unroll
    for (int ii = 0; ii < 4; ++ii) {
        int i = tid + 256*ii;
        float x = attnf[(size_t)t*HH + i] + x1f[(size_t)t*HH + i];
        v[ii] = x;
        out_res[(size_t)t*HH + i] = x;
        out_moe[(size_t)t*HH + i] = 0.0f;
        ss += x*x;
    }
#pragma unroll
    for (int off = 16; off >= 1; off >>= 1) ss += __shfl_xor(ss, off, 32);
    if ((tid & 31) == 0) sred[tid >> 5] = ss;
    __syncthreads();
    float tot = 0.f;
#pragma unroll
    for (int w = 0; w < 8; ++w) tot += sred[w];
    float sc = rsqrtf(tot / (float)HH + EPSF);
#pragma unroll
    for (int ii = 0; ii < 4; ++ii) {
        int i = tid + 256*ii;
        h2b[(size_t)t*HH + i] = f2bf(v[ii] * sc * ln2w[i]);
    }
    if (blockIdx.x == 0 && tid < EE) cnt[tid] = 0;
}

// ---------------------------------------------------------------- generic C = A(MxK) * B(NxK)^T, bf16 in f32 out
// block = 128 thr (4 waves); each wave does a 64x64 tile; block covers 128x128.
// __launch_bounds__(.,1): allow the full wave32 VGPR budget -> no spills.
__global__ __launch_bounds__(128, 1)
void k_gemm_nt(const u16* __restrict__ A, const u16* __restrict__ B,
               float* __restrict__ C, int M, int N, int Klen) {
    int lane = threadIdx.x & 31;
    int w    = threadIdx.x >> 5;
    int n0 = blockIdx.x*128 + (w & 1)*64;
    int m0 = blockIdx.y*128 + (w >> 1)*64;
    int nIdx = lane & 15, hi = lane >> 4;
    v8f acc[4][4] = {};
    // single base pointers; per-row offsets folded into immediates of the loads
    const u16* abase = A + (size_t)(m0 + nIdx) * Klen;   // + i*16*Klen
    const u16* bbase = B + (size_t)(n0 + nIdx) * Klen;   // + j*16*Klen
    const size_t rstride = (size_t)16 * Klen;
    for (int k = 0; k < Klen; k += 32) {
        v16bf a[4], b[4];
#pragma unroll
        for (int i = 0; i < 4; ++i) a[i] = load_fragA(abase + i*rstride, k, hi);
#pragma unroll
        for (int j = 0; j < 4; ++j) b[j] = load_fragB(bbase + j*rstride, k, hi);
#pragma unroll
        for (int i = 0; i < 4; ++i)
#pragma unroll
            for (int j = 0; j < 4; ++j) acc[i][j] = WMMA_BF16(a[i], b[j], acc[i][j]);
    }
    float* cbase = C + (size_t)(m0 + 8*hi) * N + n0 + nIdx;
#pragma unroll
    for (int i = 0; i < 4; ++i)
#pragma unroll
        for (int r = 0; r < 8; ++r) {
            float* crow = cbase + (size_t)(16*i + r) * N;
#pragma unroll
            for (int j = 0; j < 4; ++j) crow[16*j] = acc[i][j][r];
        }
}

// ---------------------------------------------------------------- per-head RMS + RoPE; v transposed
// grid = T, block = 32 * 24 (one wave per head/kv-head/v-head)
__global__ void k_rope(const float* __restrict__ qkvf, const int* __restrict__ positions,
                       const float* __restrict__ qnw, const float* __restrict__ knw,
                       u16* __restrict__ qb, u16* __restrict__ kb, u16* __restrict__ vTb) {
    int t = blockIdx.x;
    int wave = threadIdx.x >> 5, lane = threadIdx.x & 31;
    const float* row = qkvf + (size_t)t * QKVW;
    if (wave >= NHD + NKV) {                         // V: store transposed [NKV][D][T]
        int kv = wave - NHD - NKV;
        const float* base = row + NHD*DD + NKV*DD + kv*DD;
        vTb[((size_t)kv*DD + lane)      * TT + t] = f2bf(base[lane]);
        vTb[((size_t)kv*DD + lane + 32) * TT + t] = f2bf(base[lane + 32]);
        return;
    }
    bool isq = (wave < NHD);
    int idx = isq ? wave : wave - NHD;
    const float* base = isq ? (row + idx*DD) : (row + NHD*DD + idx*DD);
    const float* nw   = isq ? qnw : knw;
    float e0 = base[lane], e1 = base[lane + 32];
    float ss = e0*e0 + e1*e1;
#pragma unroll
    for (int off = 16; off >= 1; off >>= 1) ss += __shfl_xor(ss, off, 32);
    float sc = rsqrtf(ss / (float)DD + EPSF);
    float x0 = e0 * sc * nw[lane], x1 = e1 * sc * nw[lane + 32];
    float inv = __powf(THETAF, -(2.0f * (float)lane) / (float)DD);
    float fr = (float)positions[t] * inv;
    float sv, cv; __sincosf(fr, &sv, &cv);
    float o0 = x0*cv - x1*sv;
    float o1 = x1*cv + x0*sv;
    u16* dst = isq ? (qb + ((size_t)idx*TT + t)*DD) : (kb + ((size_t)idx*TT + t)*DD);
    dst[lane]      = f2bf(o0);
    dst[lane + 32] = f2bf(o1);
}

// ---------------------------------------------------------------- flash attention: 1 wave per (head, 16-token q tile)
__global__ __launch_bounds__(32, 1)
void k_flash(const u16* __restrict__ qb, const u16* __restrict__ kb,
             const u16* __restrict__ vT, u16* __restrict__ ob) {
    __shared__ __align__(16) u16 lds[16 * 32];
    int qt = blockIdx.x, h = blockIdx.y;
    int lane = threadIdx.x, nIdx = lane & 15, hi = lane >> 4;
    int hkv = h >> 2;                                   // NH/NKV = 4
    const u16* qrow = qb + ((size_t)h*TT + qt*16 + nIdx) * DD;
    v16bf aq[2];
    aq[0] = load_fragA(qrow, 0, hi);
    aq[1] = load_fragA(qrow, 32, hi);
    v8f O[4] = {};
    float mrow[8], lrow[8];
    int rowg[8];
#pragma unroll
    for (int r = 0; r < 8; ++r) { mrow[r] = -1e30f; lrow[r] = 0.f; rowg[r] = qt*16 + r + 8*hi; }
    const float scale = 0.125f;                         // D^-0.5
    int kend = 16 * (qt + 1);
    for (int kt0 = 0; kt0 < kend; kt0 += 32) {
        v8f S0 = {}, S1 = {};
        const u16* krow0 = kb + ((size_t)hkv*TT + kt0 + nIdx) * DD;
#pragma unroll
        for (int j = 0; j < 2; ++j) {
            S0 = WMMA_BF16(aq[j], load_fragB(krow0,          32*j, hi), S0);
            S1 = WMMA_BF16(aq[j], load_fragB(krow0 + 16*DD,  32*j, hi), S1);
        }
        float p0[8], p1[8], alpha[8];
#pragma unroll
        for (int r = 0; r < 8; ++r) {
            float s0 = S0[r] * scale; if (kt0 + nIdx      > rowg[r]) s0 = -1e30f;
            float s1 = S1[r] * scale; if (kt0 + 16 + nIdx > rowg[r]) s1 = -1e30f;
            float mx = fmaxf(s0, s1);
#pragma unroll
            for (int off = 8; off >= 1; off >>= 1) mx = fmaxf(mx, __shfl_xor(mx, off, 32));
            float mn = fmaxf(mrow[r], mx);
            alpha[r] = __expf(mrow[r] - mn);
            p0[r] = __expf(s0 - mn);
            p1[r] = __expf(s1 - mn);
            float ps = p0[r] + p1[r];
#pragma unroll
            for (int off = 8; off >= 1; off >>= 1) ps += __shfl_xor(ps, off, 32);
            lrow[r] = lrow[r] * alpha[r] + ps;
            mrow[r] = mn;
        }
#pragma unroll
        for (int j = 0; j < 4; ++j)
#pragma unroll
            for (int r = 0; r < 8; ++r) O[j][r] *= alpha[r];
        // P (D-layout) -> LDS -> A-layout fragment
#pragma unroll
        for (int r = 0; r < 8; ++r) {
            lds[(r + 8*hi)*32 + nIdx]      = f2bf(p0[r]);
            lds[(r + 8*hi)*32 + 16 + nIdx] = f2bf(p1[r]);
        }
        __syncthreads();
        FragU pf;
        pf.q[0] = *(const uint4*)(lds + nIdx*32 + 8*hi);
        pf.q[1] = *(const uint4*)(lds + nIdx*32 + 16 + 8*hi);
        __syncthreads();
        const u16* vrow = vT + ((size_t)(hkv*DD + nIdx)) * TT;
#pragma unroll
        for (int j = 0; j < 4; ++j)
            O[j] = WMMA_BF16(pf.v, load_fragB(vrow + (size_t)16*j*TT, kt0, hi), O[j]);
    }
    float inv[8];
#pragma unroll
    for (int r = 0; r < 8; ++r) inv[r] = 1.0f / lrow[r];
    u16* obase = ob + (size_t)(qt*16 + 8*hi) * (NHD*DD) + h*DD + nIdx;
#pragma unroll
    for (int j = 0; j < 4; ++j)
#pragma unroll
        for (int r = 0; r < 8; ++r)
            obase[(size_t)r * (NHD*DD) + 16*j] = f2bf(O[j][r] * inv[r]);
}

// ---------------------------------------------------------------- router: softmax + top-4 + slot claim (1 wave / token)
__global__ __launch_bounds__(32, 1)
void k_router(const u16* __restrict__ h2b, const float* __restrict__ gate_w,
              int* __restrict__ cnt, int* __restrict__ eidx, float* __restrict__ ewgt) {
    int t = blockIdx.x, lane = threadIdx.x;          // lane == expert id
    const u16*  hrow = h2b + (size_t)t * HH;
    const float* grow = gate_w + (size_t)lane * HH;
    float acc = 0.f;
    for (int i = 0; i < HH; i += 4) {
        acc += bf2f(hrow[i])   * grow[i];
        acc += bf2f(hrow[i+1]) * grow[i+1];
        acc += bf2f(hrow[i+2]) * grow[i+2];
        acc += bf2f(hrow[i+3]) * grow[i+3];
    }
    float mx = acc;
#pragma unroll
    for (int off = 16; off >= 1; off >>= 1) mx = fmaxf(mx, __shfl_xor(mx, off, 32));
    float ex = __expf(acc - mx);
    float sm = ex;
#pragma unroll
    for (int off = 16; off >= 1; off >>= 1) sm += __shfl_xor(sm, off, 32);
    float pv = ex / sm;
    int   sel[TOPK]; float selw[TOPK]; float wsum = 0.f;
#pragma unroll
    for (int kk = 0; kk < TOPK; ++kk) {
        float m = pv; int mi = lane;
#pragma unroll
        for (int off = 16; off >= 1; off >>= 1) {
            float om = __shfl_xor(m, off, 32);
            int   oi = __shfl_xor(mi, off, 32);
            if (om > m || (om == m && oi < mi)) { m = om; mi = oi; }
        }
        sel[kk] = mi; selw[kk] = m; wsum += m;
        if (lane == mi) pv = -1.0f;
    }
    if (lane < TOPK) {
        int e = sel[lane];
        float wv = selw[lane] / wsum;
        int slot = atomicAdd(&cnt[e], 1);
        eidx[e*TT + slot] = t;
        ewgt[e*TT + slot] = wv;
    }
}

// ---------------------------------------------------------------- MoE GEMM1: act = silu(h@Wg^T) * (h@Wu^T), gathered rows
// grid (II/64, TT/16, EE), block 32
__global__ __launch_bounds__(32, 1)
void k_moe_gemm1(const u16* __restrict__ h2b, const u16* __restrict__ wgu,
                 const int* __restrict__ cnt, const int* __restrict__ eidx,
                 u16* __restrict__ act) {
    int e = blockIdx.z, st = blockIdx.y, ng0 = blockIdx.x * 64;
    int cE = cnt[e];
    if (st*16 >= cE) return;
    int lane = threadIdx.x, nIdx = lane & 15, hi = lane >> 4;
    int slot = st*16 + nIdx;
    int sclamp = slot < cE ? slot : cE - 1;
    int tok = eidx[e*TT + sclamp];
    const u16* arow  = h2b + (size_t)tok * HH;
    const u16* gbase = wgu + ((size_t)e * (2*II) + ng0 + nIdx) * HH;   // + j*16*HH; u at +II*HH
    const size_t rstride = (size_t)16 * HH;
    const size_t uoff    = (size_t)II * HH;
    v8f ag[4] = {}, au[4] = {};
    for (int k = 0; k < HH; k += 32) {
        v16bf a = load_fragA(arow, k, hi);
#pragma unroll
        for (int j = 0; j < 4; ++j) {
            ag[j] = WMMA_BF16(a, load_fragB(gbase + j*rstride,        k, hi), ag[j]);
            au[j] = WMMA_BF16(a, load_fragB(gbase + uoff + j*rstride, k, hi), au[j]);
        }
    }
    u16* abase = act + ((size_t)e*TT + st*16 + 8*hi) * II + ng0 + nIdx;
#pragma unroll
    for (int j = 0; j < 4; ++j)
#pragma unroll
        for (int r = 0; r < 8; ++r) {
            float g = ag[j][r], u = au[j][r];
            float s = g / (1.0f + __expf(-g));         // silu
            abase[(size_t)r * II + 16*j] = f2bf(s * u);
        }
}

// ---------------------------------------------------------------- MoE GEMM2: moe_out += w * (act @ Wd^T), scattered
// grid (HH/64, TT/16, EE), block 32
__global__ __launch_bounds__(32, 1)
void k_moe_gemm2(const u16* __restrict__ act, const u16* __restrict__ wdn,
                 const int* __restrict__ cnt, const int* __restrict__ eidx,
                 const float* __restrict__ ewgt, float* __restrict__ moe_out) {
    int e = blockIdx.z, st = blockIdx.y, h0 = blockIdx.x * 64;
    int cE = cnt[e];
    if (st*16 >= cE) return;
    int lane = threadIdx.x, nIdx = lane & 15, hi = lane >> 4;
    const u16* arow  = act + ((size_t)e*TT + st*16 + nIdx) * II;
    const u16* bbase = wdn + ((size_t)e*HH + h0 + nIdx) * II;          // + j*16*II
    const size_t rstride = (size_t)16 * II;
    v8f acc[4] = {};
    for (int k = 0; k < II; k += 32) {
        v16bf a = load_fragA(arow, k, hi);
#pragma unroll
        for (int j = 0; j < 4; ++j) acc[j] = WMMA_BF16(a, load_fragB(bbase + j*rstride, k, hi), acc[j]);
    }
    int toks[8]; float ws[8]; bool val[8];
#pragma unroll
    for (int r = 0; r < 8; ++r) {
        int slot = st*16 + r + 8*hi;
        val[r] = slot < cE;
        toks[r] = val[r] ? eidx[e*TT + slot] : 0;
        ws[r]   = val[r] ? ewgt[e*TT + slot] : 0.f;
    }
#pragma unroll
    for (int j = 0; j < 4; ++j)
#pragma unroll
        for (int r = 0; r < 8; ++r)
            if (val[r])
                atomicAdd(&moe_out[(size_t)toks[r]*HH + h0 + 16*j + nIdx], ws[r] * acc[j][r]);
}

// ---------------------------------------------------------------- host launcher
extern "C" void kernel_launch(void* const* d_in, const int* in_sizes, int n_in,
                              void* d_out, int out_size, void* d_ws, size_t ws_size,
                              hipStream_t stream) {
    const int*   positions = (const int*)  d_in[0];
    const float* hs        = (const float*)d_in[1];
    const float* resid     = (const float*)d_in[2];
    const float* w_qkv     = (const float*)d_in[3];
    const float* w_o       = (const float*)d_in[4];
    const float* qnw       = (const float*)d_in[5];
    const float* knw       = (const float*)d_in[6];
    const float* ln1       = (const float*)d_in[7];
    const float* ln2       = (const float*)d_in[8];
    const float* gate_w    = (const float*)d_in[9];
    const float* wgu_f     = (const float*)d_in[10];
    const float* wdn_f     = (const float*)d_in[11];
    float* out_moe = (float*)d_out;
    float* out_res = (float*)d_out + (size_t)TT * HH;

    // workspace carve-up
    char* base = (char*)d_ws;
    size_t off = 0;
    auto carve = [&](size_t bytes) -> char* {
        char* p = base + off;
        off = (off + bytes + 255) & ~(size_t)255;
        return p;
    };
    u16*   wqkv_b = (u16*)  carve((size_t)QKVW * HH * 2);
    u16*   wo_b   = (u16*)  carve((size_t)HH * NHD * DD * 2);
    u16*   wgu_b  = (u16*)  carve((size_t)EE * 2 * II * HH * 2);
    u16*   wdn_b  = (u16*)  carve((size_t)EE * HH * II * 2);
    float* x1f    = (float*)carve((size_t)TT * HH * 4);
    u16*   h1b    = (u16*)  carve((size_t)TT * HH * 2);
    float* qkvf   = (float*)carve((size_t)TT * QKVW * 4);
    u16*   qb     = (u16*)  carve((size_t)NHD * TT * DD * 2);
    u16*   kb     = (u16*)  carve((size_t)NKV * TT * DD * 2);
    u16*   vTb    = (u16*)  carve((size_t)NKV * DD * TT * 2);
    u16*   ob     = (u16*)  carve((size_t)TT * NHD * DD * 2);
    float* attnf  = (float*)carve((size_t)TT * HH * 4);
    u16*   h2b    = (u16*)  carve((size_t)TT * HH * 2);
    int*   cnt    = (int*)  carve(EE * 4);
    int*   eidx   = (int*)  carve((size_t)EE * TT * 4);
    float* ewgt   = (float*)carve((size_t)EE * TT * 4);
    u16*   act    = (u16*)  carve((size_t)EE * TT * II * 2);
    (void)ws_size; (void)in_sizes; (void)n_in; (void)out_size;

    // 1) weights -> bf16
    k_f2bf<<<2048, 256, 0, stream>>>(w_qkv, wqkv_b, QKVW * HH);
    k_f2bf<<<2048, 256, 0, stream>>>(w_o,   wo_b,   HH * NHD * DD);
    k_f2bf<<<8192, 256, 0, stream>>>(wgu_f, wgu_b,  EE * 2 * II * HH);
    k_f2bf<<<8192, 256, 0, stream>>>(wdn_f, wdn_b,  EE * HH * II);

    // 2) residual add + RMSNorm
    k_prep1<<<TT, 256, 0, stream>>>(hs, resid, ln1, x1f, h1b);

    // 3) QKV GEMM (1024x1024x1536)
    k_gemm_nt<<<dim3(QKVW/128, TT/128), 128, 0, stream>>>(h1b, wqkv_b, qkvf, TT, QKVW, HH);

    // 4) per-head q/k RMSNorm + RoPE, v transpose
    k_rope<<<TT, 32 * (NHD + 2*NKV), 0, stream>>>(qkvf, positions, qnw, knw, qb, kb, vTb);

    // 5) flash attention
    k_flash<<<dim3(TT/16, NHD), 32, 0, stream>>>(qb, kb, vTb, ob);

    // 6) output projection GEMM
    k_gemm_nt<<<dim3(HH/128, TT/128), 128, 0, stream>>>(ob, wo_b, attnf, TT, HH, HH);

    // 7) residual out + RMSNorm + zero moe_out + zero router counts
    k_prep2<<<TT, 256, 0, stream>>>(attnf, x1f, ln2, out_moe, out_res, h2b, cnt);

    // 8) router (softmax + top-4 + slot claim)
    k_router<<<TT, 32, 0, stream>>>(h2b, gate_w, cnt, eidx, ewgt);

    // 9) MoE expert GEMMs (sparse, gathered/scattered)
    k_moe_gemm1<<<dim3(II/64, TT/16, EE), 32, 0, stream>>>(h2b, wgu_b, cnt, eidx, act);
    k_moe_gemm2<<<dim3(HH/64, TT/16, EE), 32, 0, stream>>>(act, wdn_b, cnt, eidx, ewgt, out_moe);
}
